// Fuser_32564442038514
// MI455X (gfx1250) — matrix operand order repaired and checked by reading, not compiled, not run on hardware
//
#include <hip/hip_runtime.h>

// Problem dimensions (match the reference's setup_inputs()).
#define N_S  4096
#define N_K  8192
#define HID  1024
#define SENT 1024

typedef __attribute__((ext_vector_type(16))) __bf16 v16bf;
typedef __attribute__((ext_vector_type(8)))  float  v8f;

// f32 -> bf16 round-to-nearest-even (inputs finite).
__device__ inline unsigned short f2bf(float f) {
    unsigned int u = __builtin_bit_cast(unsigned int, f);
    unsigned int r = u + 0x7FFFu + ((u >> 16) & 1u);
    return (unsigned short)(r >> 16);
}

union FragU { uint4 q[2]; unsigned int u[8]; v16bf v; };

// ---------------------------------------------------------------------------
// CDNA5 async global->LDS copy (ASYNCcnt path), 16B per lane.
//   LDS byte address from a VGPR; global address = SGPR base + VGPR byte offset
// ---------------------------------------------------------------------------
__device__ inline void async_copy_b128(unsigned ldsByteOff,
                                       const unsigned short* gbase,
                                       unsigned gByteOff) {
    asm volatile("global_load_async_to_lds_b128 %0, %1, %2"
                 :: "v"(ldsByteOff), "v"(gByteOff), "s"(gbase)
                 : "memory");
}
__device__ inline void wait_async0() {
    asm volatile("s_wait_asynccnt 0" ::: "memory");
}

// ---------------------------------------------------------------------------
// bf16 WMMA GEMM:  C[M,N] = A[M,Kd] @ Bt[N,Kd]^T (+ bias)
// A, Bt pre-converted bf16. Double-buffered async LDS staging.
// M,N multiples of 128; Kd multiple of 64.
// ---------------------------------------------------------------------------
constexpr int BM = 128, BN = 128, BK = 64;

template <int OUT_BF16, int HAS_BIAS>
__global__ __launch_bounds__(256)
void gemm_bf16(const unsigned short* __restrict__ A,
               const unsigned short* __restrict__ Bt,
               const float* __restrict__ bias,
               void* __restrict__ Cout,
               int M, int N, int Kd)
{
    __shared__ unsigned short sA[2][BM * BK];
    __shared__ unsigned short sB[2][BN * BK];

    const int tid   = threadIdx.x;
    const int lane  = tid & 31;
    const int wave  = tid >> 5;
    const int waveM = (wave >> 2) * 64;   // 2 (M) x 4 (N) wave grid over 128x128
    const int waveN = (wave & 3)  * 32;
    const int mBase = blockIdx.y * BM;
    const int nBase = blockIdx.x * BN;
    const int r     = lane & 15;
    const int half  = lane >> 4;

    const unsigned ldsA0 = (unsigned)(unsigned long long)(void*)&sA[0][0];
    const unsigned ldsB0 = (unsigned)(unsigned long long)(void*)&sB[0][0];

    v8f acc[4][2] = {};

    const int nT = Kd / BK;

    // Issue one stage of async copies: 128 rows x 64 k of A and Bt (16 KB each).
    // 1024 chunks of 16B per array; 4 per thread.
    auto issue = [&](int buf, int k0) {
        #pragma unroll
        for (int i = 0; i < 4; ++i) {
            int c   = tid + i * 256;
            int row = c >> 3;
            int kc  = (c & 7) * 8;
            unsigned lOff   = (unsigned)(row * BK + kc) * 2u;
            unsigned bufOff = (unsigned)(buf * BM * BK) * 2u;
            unsigned gA = (unsigned)((mBase + row) * Kd + k0 + kc) * 2u;
            async_copy_b128(ldsA0 + bufOff + lOff, A, gA);
            unsigned gB = (unsigned)((nBase + row) * Kd + k0 + kc) * 2u;
            async_copy_b128(ldsB0 + bufOff + lOff, Bt, gB);
        }
    };

    issue(0, 0);
    for (int t = 0; t < nT; ++t) {
        wait_async0();          // this wave's copies for buffer t&1 have landed
        __syncthreads();        // all waves' copies landed; all prior reads done
        if (t + 1 < nT) issue((t + 1) & 1, (t + 1) * BK);

        const unsigned short* a0 = &sA[t & 1][0];
        const unsigned short* b0 = &sB[t & 1][0];

        #pragma unroll
        for (int ks = 0; ks < 2; ++ks) {    // two K=32 WMMA steps per BK=64 stage
            FragU a[4], b[2];
            #pragma unroll
            for (int tm = 0; tm < 4; ++tm) {
                // A 16x32 bf16 frag: lane half splits K {0..7,16..23}/{8..15,24..31}
                const unsigned short* p = a0 + (waveM + tm * 16 + r) * BK + ks * 32 + half * 8;
                a[tm].q[0] = *(const uint4*)p;
                a[tm].q[1] = *(const uint4*)(p + 16);
            }
            #pragma unroll
            for (int tn = 0; tn < 2; ++tn) {
                // B 32x16 bf16 frag: lane half splits K {0..7,16..23}/{8..15,24..31}
                const unsigned short* p = b0 + (waveN + tn * 16 + r) * BK + ks * 32 + half * 8;
                b[tn].q[0] = *(const uint4*)p;
                b[tn].q[1] = *(const uint4*)(p + 16);
            }
            #pragma unroll
            for (int tm = 0; tm < 4; ++tm)
                #pragma unroll
                for (int tn = 0; tn < 2; ++tn)
                    acc[tm][tn] = __builtin_amdgcn_wmma_f32_16x16x32_bf16(
                        false, a[tm].v, false, b[tn].v,
                        (short)0, acc[tm][tn], false, false);
        }
        __syncthreads();
    }

    // C/D layout: lanes 0-15 -> M=v, lanes 16-31 -> M=v+8; N = lane&15.
    #pragma unroll
    for (int tm = 0; tm < 4; ++tm) {
        #pragma unroll
        for (int tn = 0; tn < 2; ++tn) {
            int col  = nBase + waveN + tn * 16 + r;
            float bv = HAS_BIAS ? bias[col] : 0.0f;
            #pragma unroll
            for (int v = 0; v < 8; ++v) {
                int row   = mBase + waveM + tm * 16 + v + 8 * half;
                float val = acc[tm][tn][v] + bv;
                if (OUT_BF16)
                    ((unsigned short*)Cout)[(size_t)row * N + col] = f2bf(val);
                else
                    ((float*)Cout)[(size_t)row * N + col] = val;
            }
        }
    }
}

// ---------------------------------------------------------------------------
// Streaming f32 -> bf16 conversion (n multiple of 2048).
// ---------------------------------------------------------------------------
__global__ __launch_bounds__(256)
void convert_bf16(const float* __restrict__ src, unsigned short* __restrict__ dst, int n)
{
    int i = (blockIdx.x * 256 + threadIdx.x) * 8;
    float4 f0 = *(const float4*)(src + i);
    float4 f1 = *(const float4*)(src + i + 4);
    uint4 o;
    o.x = (unsigned)f2bf(f0.x) | ((unsigned)f2bf(f0.y) << 16);
    o.y = (unsigned)f2bf(f0.z) | ((unsigned)f2bf(f0.w) << 16);
    o.z = (unsigned)f2bf(f1.x) | ((unsigned)f2bf(f1.y) << 16);
    o.w = (unsigned)f2bf(f1.z) | ((unsigned)f2bf(f1.w) << 16);
    *(uint4*)(dst + i) = o;
}

// ---------------------------------------------------------------------------
// Tiled f32 [R,C] -> bf16 [C,R] transpose-convert. R, C multiples of 32.
// ---------------------------------------------------------------------------
__global__ __launch_bounds__(256)
void transpose_bf16(const float* __restrict__ src, unsigned short* __restrict__ dst,
                    int R, int C)
{
    __shared__ unsigned short tile[32][33];
    int c0 = blockIdx.x * 32, r0 = blockIdx.y * 32;
    int lr = threadIdx.x >> 5;          // 0..7
    int lc = threadIdx.x & 31;          // 0..31
    #pragma unroll
    for (int i = 0; i < 4; ++i) {
        int rr = lr + i * 8;
        tile[lc][rr] = f2bf(src[(size_t)(r0 + rr) * C + (c0 + lc)]);
    }
    __syncthreads();
    #pragma unroll
    for (int i = 0; i < 4; ++i) {
        int rr = lr + i * 8;
        dst[(size_t)(c0 + rr) * R + (r0 + lc)] = tile[rr][lc];
    }
}

// ---------------------------------------------------------------------------
// Split-row softmax over dim 0 of attn[rows, cols] (column-wise).
// part1: per (colBlock, rowChunk) partial max / exp-sum   (grid 32 x 16)
// part2: combine 16 chunk partials per column             (grid 32)
// part3: normalize, write f32 output + bf16 mirror        (grid 32 x 16)
// ---------------------------------------------------------------------------
__global__ __launch_bounds__(256)
void softmax_part1(const float* __restrict__ attn,
                   float* __restrict__ partM, float* __restrict__ partL,
                   int rows, int cols, int rowsPerChunk)
{
    int c  = blockIdx.x * 256 + threadIdx.x;     // cols multiple of 256
    int r0 = blockIdx.y * rowsPerChunk;
    float m = -3.402823466e38f;
    for (int rr = r0; rr < r0 + rowsPerChunk; ++rr)
        m = fmaxf(m, attn[(size_t)rr * cols + c]);
    float s = 0.0f;
    for (int rr = r0; rr < r0 + rowsPerChunk; ++rr)
        s += __expf(attn[(size_t)rr * cols + c] - m);
    partM[(size_t)blockIdx.y * cols + c] = m;
    partL[(size_t)blockIdx.y * cols + c] = s;
}

__global__ __launch_bounds__(256)
void softmax_part2(const float* __restrict__ partM, const float* __restrict__ partL,
                   float* __restrict__ colM, float* __restrict__ colInvL,
                   int cols, int nChunks)
{
    int c = blockIdx.x * 256 + threadIdx.x;
    float M = -3.402823466e38f;
    for (int i = 0; i < nChunks; ++i)
        M = fmaxf(M, partM[(size_t)i * cols + c]);
    float L = 0.0f;
    for (int i = 0; i < nChunks; ++i)
        L += partL[(size_t)i * cols + c] * __expf(partM[(size_t)i * cols + c] - M);
    colM[c]    = M;
    colInvL[c] = 1.0f / L;
}

__global__ __launch_bounds__(256)
void softmax_part3(float* __restrict__ attn, unsigned short* __restrict__ attnBf,
                   const float* __restrict__ colM, const float* __restrict__ colInvL,
                   int rows, int cols, int rowsPerChunk)
{
    int c  = blockIdx.x * 256 + threadIdx.x;
    int r0 = blockIdx.y * rowsPerChunk;
    float M   = colM[c];
    float inv = colInvL[c];
    for (int rr = r0; rr < r0 + rowsPerChunk; ++rr) {
        size_t i = (size_t)rr * cols + c;
        float v  = __expf(attn[i] - M) * inv;
        attn[i]   = v;
        attnBf[i] = f2bf(v);
    }
}

// ---------------------------------------------------------------------------

extern "C" void kernel_launch(void* const* d_in, const int* in_sizes, int n_in,
                              void* d_out, int out_size, void* d_ws, size_t ws_size,
                              hipStream_t stream)
{
    const float* sentences = (const float*)d_in[0];   // [N_S, SENT]
    const float* knowledge = (const float*)d_in[1];   // [N_K, 3*HID]
    const float* Ws        = (const float*)d_in[2];   // [SENT, HID]
    const float* bs        = (const float*)d_in[3];   // [HID]
    const float* Wk        = (const float*)d_in[4];   // [3*HID, HID]
    const float* bk        = (const float*)d_in[5];   // [HID]

    float* fused = (float*)d_out;                                  // [N_S, 3*HID]
    float* attn  = fused + (size_t)N_S * (3 * HID);                // [N_S, N_K]

    // Workspace layout (bf16 elements first, then f32 reduction scratch).
    unsigned short* sentB = (unsigned short*)d_ws;                 // [N_S, SENT]
    unsigned short* WsT   = sentB + (size_t)N_S * SENT;            // [HID, SENT]
    unsigned short* knowB = WsT   + (size_t)HID * SENT;            // [N_K, 3*HID]
    unsigned short* WkT   = knowB + (size_t)N_K * 3 * HID;         // [HID, 3*HID]
    unsigned short* knowT = WkT   + (size_t)HID * 3 * HID;         // [3*HID, N_K]
    unsigned short* Sbf   = knowT + (size_t)3 * HID * N_K;         // [N_S, HID]
    unsigned short* Kbf   = Sbf   + (size_t)N_S * HID;             // [N_K, HID]
    unsigned short* attnB = Kbf   + (size_t)N_K * HID;             // [N_S, N_K]

    const int ROW_CHUNK = 256;                                     // 16 chunks of N_S
    const int N_CHUNK   = N_S / ROW_CHUNK;
    float* partM   = (float*)(attnB + (size_t)N_S * N_K);          // [N_CHUNK, N_K]
    float* partL   = partM + (size_t)N_CHUNK * N_K;                // [N_CHUNK, N_K]
    float* colM    = partL + (size_t)N_CHUNK * N_K;                // [N_K]
    float* colInvL = colM + N_K;                                   // [N_K]

    dim3 blk(256);

    // One-time f32 -> bf16 conversions / transposes (streaming, ~15 us of HBM time).
    convert_bf16<<<(N_S * SENT) / 2048, blk, 0, stream>>>(sentences, sentB, N_S * SENT);
    transpose_bf16<<<dim3(HID / 32, SENT / 32), blk, 0, stream>>>(Ws, WsT, SENT, HID);
    convert_bf16<<<(N_K * 3 * HID) / 2048, blk, 0, stream>>>(knowledge, knowB, N_K * 3 * HID);
    transpose_bf16<<<dim3(HID / 32, (3 * HID) / 32), blk, 0, stream>>>(Wk, WkT, 3 * HID, HID);
    transpose_bf16<<<dim3((3 * HID) / 32, N_K / 32), blk, 0, stream>>>(knowledge, knowT, N_K, 3 * HID);

    // S = sentences @ Ws + bs  (bf16 out)
    gemm_bf16<1, 1><<<dim3(HID / BN, N_S / BM), blk, 0, stream>>>(
        sentB, WsT, bs, Sbf, N_S, HID, SENT);

    // K = knowledge @ Wk + bk  (bf16 out)
    gemm_bf16<1, 1><<<dim3(HID / BN, N_K / BM), blk, 0, stream>>>(
        knowB, WkT, bk, Kbf, N_K, HID, 3 * HID);

    // scores = S @ K^T  -> f32, straight into the attentions output region
    gemm_bf16<0, 0><<<dim3(N_K / BN, N_S / BM), blk, 0, stream>>>(
        Sbf, Kbf, nullptr, attn, N_S, N_K, HID);

    // attentions = softmax(scores, axis=0), split-row reduction for full parallelism
    softmax_part1<<<dim3(N_K / 256, N_CHUNK), blk, 0, stream>>>(
        attn, partM, partL, N_S, N_K, ROW_CHUNK);
    softmax_part2<<<N_K / 256, blk, 0, stream>>>(
        partM, partL, colM, colInvL, N_K, N_CHUNK);
    softmax_part3<<<dim3(N_K / 256, N_CHUNK), blk, 0, stream>>>(
        attn, attnB, colM, colInvL, N_S, N_K, ROW_CHUNK);

    // fused = attentions @ knowledge  (f32 out)
    gemm_bf16<0, 0><<<dim3((3 * HID) / BN, N_S / BM), blk, 0, stream>>>(
        attnB, knowT, nullptr, fused, N_S, 3 * HID, N_K);
}
